// AlignedNet_51075751084587
// MI455X (gfx1250) — compile-verified, matrix operand order checked by reading, not run on hardware
//
#include <hip/hip_runtime.h>
#include <hip/hip_bf16.h>

// ---------------------------------------------------------------------------
// DCNv2 alignment network for MI455X (gfx1250), wave32 + WMMA f16 (f32 acc).
// All convs lowered to implicit GEMM: M=Cout, N=9216 pixels, K=Cin*9.
// Per-wave tile: M=64 (4 WMMA accumulators) x N=16, so each staged B tile
// (one 32B vector load per lane) feeds 4 v_wmma_f32_16x16x32_f16.
// ---------------------------------------------------------------------------

typedef _Float16 v16h  __attribute__((ext_vector_type(16)));
typedef _Float16 v16hu __attribute__((ext_vector_type(16), aligned(2)));  // unaligned load
typedef _Float16 v8h   __attribute__((ext_vector_type(8)));
typedef float    v8f   __attribute__((ext_vector_type(8)));

#define HP   96
#define WPX  96
#define NPIX (HP * WPX)        // 9216 pixels per image
#define NB   2                 // batch
#define WAVES 4                // waves per block in the GEMM kernel
#define TN   16                // N-tile per wave
#define TM   64                // M-tile per wave (4 x 16)

// ---------------------------------------------------------------------------
// Elementwise helpers
// ---------------------------------------------------------------------------
__global__ void cvt_f32_to_f16_kernel(const float* __restrict__ src,
                                      _Float16* __restrict__ dst, int n) {
    for (int i = blockIdx.x * blockDim.x + threadIdx.x; i < n;
         i += gridDim.x * blockDim.x)
        dst[i] = (_Float16)src[i];
}

__global__ void prep_kernel(const float* __restrict__ R1,
                            const float* __restrict__ Q0,
                            _Float16* __restrict__ diffh,
                            _Float16* __restrict__ q0h, int n) {
    for (int i = blockIdx.x * blockDim.x + threadIdx.x; i < n;
         i += gridDim.x * blockDim.x) {
        float r = R1[i], q = Q0[i];
        diffh[i] = (_Float16)(r - q);
        q0h[i]   = (_Float16)q;
    }
}

__device__ __forceinline__ v16h pack16(v8h lo, v8h hi) {
    v16h r;
#pragma unroll
    for (int i = 0; i < 8; ++i) { r[i] = lo[i]; r[8 + i] = hi[i]; }
    return r;
}

// A-fragment: 16x32 f16, ISA layout (lanes 0-15: K 0-7 / 16-23,
// lanes 16-31: K 8-15 / 24-31).  Row pointer is pre-offset by (hi?8:0).
__device__ __forceinline__ v16h loadA(const _Float16* Arow, int k0) {
    const v8h* ap = (const v8h*)(Arow + k0);
    return pack16(ap[0], ap[2]);      // +0 halves and +16 halves
}

// ---------------------------------------------------------------------------
// Generic WMMA GEMM conv kernel.
//   mode 0: implicit im2col from single f16 feature map [b][Cin][NPIX]
//   mode 1: implicit im2col from concat(inA(128ch), inB(128ch))
//   mode 2: direct K x NPIX f16 matrix (pre-materialized deformable im2col)
// Weights Wf: f16 row-major [Cout][Ktot].  Output: optional f32 and/or f16,
// both laid out [b][Cout][NPIX], with bias + optional ReLU.
// Grid: x = NPIX/(WAVES*TN)=144, y = ceil(Cout/64), z = batch.  Block = 128.
// ---------------------------------------------------------------------------
__global__ __launch_bounds__(128)
void wmma_conv_gemm(const _Float16* __restrict__ inA,
                    const _Float16* __restrict__ inB,
                    const _Float16* __restrict__ Wf,
                    const float* __restrict__ bias,
                    float* __restrict__ outF32,
                    _Float16* __restrict__ outF16,
                    int Cout, int Ktot, int mode, int relu) {
    __shared__ __align__(16) _Float16 lds[WAVES][16][32];  // [wave][n][k]

    const int lane  = threadIdx.x & 31;
    const int wave  = threadIdx.x >> 5;
    const int b     = blockIdx.z;
    const int mbase = blockIdx.y * TM;
    const int pbase = (blockIdx.x * WAVES + wave) * TN;

    const int  laneN = lane & 15;
    const bool hi    = lane >= 16;

    // A row pointers for the 4 M-subtiles; OOB rows clamp to a valid row
    // (their D rows are masked at the store, so garbage accumulation is fine).
    const _Float16* Arow0; const _Float16* Arow1;
    const _Float16* Arow2; const _Float16* Arow3;
    {
        int a0 = mbase + 0  + laneN; if (a0 >= Cout) a0 = Cout - 1;
        int a1 = mbase + 16 + laneN; if (a1 >= Cout) a1 = Cout - 1;
        int a2 = mbase + 32 + laneN; if (a2 >= Cout) a2 = Cout - 1;
        int a3 = mbase + 48 + laneN; if (a3 >= Cout) a3 = Cout - 1;
        const int ho = hi ? 8 : 0;
        Arow0 = Wf + (size_t)a0 * Ktot + ho;
        Arow1 = Wf + (size_t)a1 * Ktot + ho;
        Arow2 = Wf + (size_t)a2 * Ktot + ho;
        Arow3 = Wf + (size_t)a3 * Ktot + ho;
    }

    const int CinA = (mode == 1) ? 128 : (Ktot / 9);
    const int ph   = pbase / WPX;          // tile lies within one image row
    const int pw   = pbase - ph * WPX;

    // Incrementally-maintained decomposition of k = k0 + lane into
    // c = k/9, r = k%9 (k advances by 32 = 3*9 + 5 each iteration).
    int cchan = lane / 9;
    int rk    = lane - cchan * 9;

    // mode 2: contiguous row pointer, advanced by 32 rows per iteration
    const _Float16* direct = inA + (size_t)b * Ktot * NPIX
                                 + (size_t)lane * NPIX + pbase;

    v8f acc0 = {}, acc1 = {}, acc2 = {}, acc3 = {};

    for (int k0 = 0; k0 < Ktot; k0 += 32) {
        __syncthreads();
        // ---- stage B tile (32 k-rows x 16 pixels): one k-row per lane -----
        {
            v16h row;
            if (mode == 2) {
                row = *(const v16hu*)direct;        // contiguous, 32B aligned
                direct += (size_t)32 * NPIX;
            } else {
                int ky  = rk / 3 - 1;
                int kx  = rk - (rk / 3) * 3 - 1;
                int hh  = ph + ky;
                int wss = pw + kx;                  // -1 .. 81
                if ((unsigned)hh < (unsigned)HP) {
                    const _Float16* basep;
                    if (mode == 1 && cchan >= 128)
                        basep = inB + ((size_t)b * 128 + (cchan - 128)) * NPIX;
                    else
                        basep = inA + ((size_t)b * CinA + cchan) * NPIX;
                    row = *(const v16hu*)(basep + hh * WPX + wss);
                    if (wss < 0)         row[0]  = (_Float16)0.0f;
                    if (wss + 15 >= WPX) row[15] = (_Float16)0.0f;
                } else {
#pragma unroll
                    for (int i = 0; i < 16; ++i) row[i] = (_Float16)0.0f;
                }
                // advance k by 32: c += 3 or 4, r = (r+5) % 9
                int r5 = rk + 5;
                if (r5 >= 9) { rk = r5 - 9; cchan += 4; }
                else         { rk = r5;     cchan += 3; }
            }
#pragma unroll
            for (int n = 0; n < 16; ++n) lds[wave][n][lane] = row[n];
        }
        __syncthreads();

        if (k0 + 32 < Ktot)
            __builtin_prefetch((const void*)(Arow0 + k0 + 32), 0, 1);

        // ---- hoist all fragment loads ahead of the WMMA block -------------
        v16h a0 = loadA(Arow0, k0);
        v16h a1 = loadA(Arow1, k0);
        v16h a2 = loadA(Arow2, k0);
        v16h a3 = loadA(Arow3, k0);
        const v8h* bp = (const v8h*)(&lds[wave][laneN][hi ? 8 : 0]);
        v16h bfrag = pack16(bp[0], bp[2]);

        // ---- 4 WMMAs share one B fragment, issued back-to-back ------------
        acc0 = __builtin_amdgcn_wmma_f32_16x16x32_f16(
            false, a0, false, bfrag, (short)0, acc0, false, false);
        acc1 = __builtin_amdgcn_wmma_f32_16x16x32_f16(
            false, a1, false, bfrag, (short)0, acc1, false, false);
        acc2 = __builtin_amdgcn_wmma_f32_16x16x32_f16(
            false, a2, false, bfrag, (short)0, acc2, false, false);
        acc3 = __builtin_amdgcn_wmma_f32_16x16x32_f16(
            false, a3, false, bfrag, (short)0, acc3, false, false);
    }

    // ---- store C/D: VGPR r -> M = sub*16 + r (+8 for hi lanes), N = laneN -
    const int n = pbase + laneN;
#pragma unroll
    for (int j = 0; j < 4; ++j) {
        v8f a = (j == 0) ? acc0 : (j == 1) ? acc1 : (j == 2) ? acc2 : acc3;
#pragma unroll
        for (int r = 0; r < 8; ++r) {
            int m = mbase + j * 16 + r + (hi ? 8 : 0);
            if (m < Cout) {
                float v = a[r] + bias[m];
                if (relu) v = fmaxf(v, 0.0f);
                size_t oidx = ((size_t)b * Cout + m) * NPIX + n;
                if (outF32) outF32[oidx] = v;
                if (outF16) outF16[oidx] = (_Float16)v;
            }
        }
    }
}

// ---------------------------------------------------------------------------
// Build deformable im2col matrix v : [b][(g*16+c)*9+k][NPIX] in f16.
// om layout [b][216][NPIX]: ch 0..143 = offsets (dy/dx interleaved per group),
// ch 144..215 = mask logits.  One thread per (b,g,k,p); loops c = 0..15.
// ---------------------------------------------------------------------------
__global__ void build_dcn_v_kernel(const float* __restrict__ R1,
                                   const float* __restrict__ om,
                                   _Float16* __restrict__ v) {
    long tid   = (long)blockIdx.x * blockDim.x + threadIdx.x;
    long total = (long)NB * 8 * 9 * NPIX;
    if (tid >= total) return;

    int p = (int)(tid % NPIX);  long t = tid / NPIX;
    int k = (int)(t % 9);       t /= 9;
    int g = (int)(t % 8);
    int b = (int)(t / 8);

    int h = p / WPX, w = p - (p / WPX) * WPX;

    const float* omb = om + (size_t)b * 216 * NPIX;
    float dy = omb[(size_t)(g * 18 + 2 * k)     * NPIX + p];
    float dx = omb[(size_t)(g * 18 + 2 * k + 1) * NPIX + p];
    float ml = omb[(size_t)(144 + g * 9 + k)    * NPIX + p];
    float m  = 1.0f / (1.0f + __expf(-ml));

    int ky = k / 3, kx = k - (k / 3) * 3;
    float py = dy + (float)(ky + h - 1);
    float px = dx + (float)(kx + w - 1);

    float y0f = floorf(py), x0f = floorf(px);
    float ly = py - y0f, lx = px - x0f;
    float y1f = y0f + 1.0f, x1f = x0f + 1.0f;

    bool vy0 = (y0f >= 0.0f) && (y0f <= 95.0f);
    bool vy1 = (y1f >= 0.0f) && (y1f <= 95.0f);
    bool vx0 = (x0f >= 0.0f) && (x0f <= 95.0f);
    bool vx1 = (x1f >= 0.0f) && (x1f <= 95.0f);

    int yc0 = (int)fminf(fmaxf(y0f, 0.0f), 95.0f);
    int yc1 = (int)fminf(fmaxf(y1f, 0.0f), 95.0f);
    int xc0 = (int)fminf(fmaxf(x0f, 0.0f), 95.0f);
    int xc1 = (int)fminf(fmaxf(x1f, 0.0f), 95.0f);

    float w00 = (vy0 && vx0) ? (1.0f - ly) * (1.0f - lx) : 0.0f;
    float w01 = (vy0 && vx1) ? (1.0f - ly) * lx          : 0.0f;
    float w10 = (vy1 && vx0) ? ly * (1.0f - lx)          : 0.0f;
    float w11 = (vy1 && vx1) ? ly * lx                   : 0.0f;

    int i00 = yc0 * WPX + xc0, i01 = yc0 * WPX + xc1;
    int i10 = yc1 * WPX + xc0, i11 = yc1 * WPX + xc1;

    const float* xb = R1 + ((size_t)b * 128 + g * 16) * NPIX;
#pragma unroll 4
    for (int c = 0; c < 16; ++c) {
        const float* xc = xb + (size_t)c * NPIX;
        float s = w00 * xc[i00] + w01 * xc[i01] + w10 * xc[i10] + w11 * xc[i11];
        v[((size_t)b * 1152 + (size_t)((g * 16 + c) * 9 + k)) * NPIX + p] =
            (_Float16)(s * m);
    }
}

// ---------------------------------------------------------------------------
// Host-side launch
// ---------------------------------------------------------------------------
extern "C" void kernel_launch(void* const* d_in, const int* in_sizes, int n_in,
                              void* d_out, int out_size, void* d_ws, size_t ws_size,
                              hipStream_t stream) {
    (void)in_sizes; (void)n_in; (void)out_size; (void)ws_size;
    const float* R1   = (const float*)d_in[0];
    const float* Q0   = (const float*)d_in[1];
    const float* w1   = (const float*)d_in[2];
    const float* b1   = (const float*)d_in[3];
    const float* w2   = (const float*)d_in[4];
    const float* b2   = (const float*)d_in[5];
    const float* wom  = (const float*)d_in[6];
    const float* bom  = (const float*)d_in[7];
    const float* wdcn = (const float*)d_in[8];
    const float* bdcn = (const float*)d_in[9];
    const float* wrq  = (const float*)d_in[10];
    const float* brq  = (const float*)d_in[11];

    char*  ws  = (char*)d_ws;
    size_t off = 0;
    auto carve = [&](size_t bytes) -> char* {
        char* p = ws + off;
        off = (off + bytes + 255) & ~(size_t)255;
        return p;
    };

    const size_t nmap = (size_t)NB * 128 * NPIX;          // 2,359,296 elems

    _Float16* w1h   = (_Float16*)carve(147456u * 2);      // 128*1152
    _Float16* w2h   = (_Float16*)carve(147456u * 2);
    _Float16* womh  = (_Float16*)carve(248832u * 2);      // 216*1152
    _Float16* wdcnh = (_Float16*)carve(147456u * 2);
    _Float16* wrqh  = (_Float16*)carve(294912u * 2);      // 128*2304
    _Float16* diffh = (_Float16*)carve(nmap * 2);
    _Float16* q0h   = (_Float16*)carve(nmap * 2);
    _Float16* off1h = (_Float16*)carve(nmap * 2);
    _Float16* off2h = (_Float16*)carve(nmap * 2);
    float*    omf   = (float*)   carve((size_t)NB * 216 * NPIX * 4);
    _Float16* vh    = (_Float16*)carve((size_t)NB * 1152 * NPIX * 2);
    _Float16* feah  = (_Float16*)carve(nmap * 2);
    (void)carve(256);   // trailing pad: edge vector loads may over-read 2B

    // --- weight conversion f32 -> f16 (flat OIHW == [Cout][Cin*9]) ---------
    cvt_f32_to_f16_kernel<<<288, 256, 0, stream>>>(w1,   w1h,   147456);
    cvt_f32_to_f16_kernel<<<288, 256, 0, stream>>>(w2,   w2h,   147456);
    cvt_f32_to_f16_kernel<<<486, 256, 0, stream>>>(wom,  womh,  248832);
    cvt_f32_to_f16_kernel<<<288, 256, 0, stream>>>(wdcn, wdcnh, 147456);
    cvt_f32_to_f16_kernel<<<576, 256, 0, stream>>>(wrq,  wrqh,  294912);

    // --- diff = R1 - Q0 (f16) and Q0 (f16) ---------------------------------
    prep_kernel<<<4608, 256, 0, stream>>>(R1, Q0, diffh, q0h, (int)nmap);

    dim3 blk(128);
    dim3 grid2(NPIX / (WAVES * TN), 2, NB);   // Cout = 128 (2 x 64 M-tiles)
    dim3 grid4(NPIX / (WAVES * TN), 4, NB);   // Cout = 216 (4 x 64, guarded)

    // conv1: relu(conv(diff))            128 -> 128, K=1152
    wmma_conv_gemm<<<grid2, blk, 0, stream>>>(diffh, nullptr, w1h, b1,
                                              nullptr, off1h, 128, 1152, 0, 1);
    // conv2: relu(conv(off1))            128 -> 128
    wmma_conv_gemm<<<grid2, blk, 0, stream>>>(off1h, nullptr, w2h, b2,
                                              nullptr, off2h, 128, 1152, 0, 1);
    // conv_om: conv(off2)                128 -> 216 (offsets + mask, f32)
    wmma_conv_gemm<<<grid4, blk, 0, stream>>>(off2h, nullptr, womh, bom,
                                              omf, nullptr, 216, 1152, 0, 0);
    // deformable gather -> im2col matrix v (f16)
    {
        long total = (long)NB * 8 * 9 * NPIX;   // 1,327,104
        int  blocks = (int)((total + 255) / 256);
        build_dcn_v_kernel<<<blocks, 256, 0, stream>>>(R1, omf, vh);
    }
    // dcn GEMM: fea = relu(Wdcn @ v + b)  128 x 1152 x 9216
    wmma_conv_gemm<<<grid2, blk, 0, stream>>>(vh, nullptr, wdcnh, bdcn,
                                              nullptr, feah, 128, 1152, 2, 1);
    // final: relu(conv(concat(fea, Q0)))  256 -> 128, K=2304
    wmma_conv_gemm<<<grid2, blk, 0, stream>>>(feah, q0h, wrqh, brq,
                                              (float*)d_out, nullptr,
                                              128, 2304, 1, 1);
}
